// HadLinear_32676111188303
// MI455X (gfx1250) — matrix operand order repaired and checked by reading, not compiled, not run on hardware
//
#include <hip/hip_runtime.h>
#include <stdint.h>

// ---------- types ----------
typedef __attribute__((ext_vector_type(16))) __bf16 v16bf;
typedef __attribute__((ext_vector_type(8)))  float  v8f;
typedef int vint4 __attribute__((vector_size(16)));   // b128 payload type for async-LDS builtin

#define AS1 __attribute__((address_space(1)))
#define AS3 __attribute__((address_space(3)))

#define BS      128      // quantization block size
#define K_DIM   4096
#define N_DIM   4096

// GEMM tiling: block 128(M) x 256(N), K-tile 64, 8 wave32s each owning 64x64
#define BM      128
#define BN      256
#define BK      64
#define APITCH  72       // LDS row pitch in bf16 elements (144B, 16B-aligned, bank-spread)

#if defined(__gfx1250__) && \
    __has_builtin(__builtin_amdgcn_global_load_async_to_lds_b128) && \
    __has_builtin(__builtin_amdgcn_s_wait_asynccnt)
#define HAS_ASYNC 1
#define NBUF 2
#else
#define HAS_ASYNC 0
#define NBUF 1
#endif

// NF4-style Gaussian quantile centroids: Phi^-1((i+0.5)/16)
__device__ __constant__ float kCentroids[16] = {
    -1.8627319f, -1.3180109f, -1.0099893f, -0.7764218f,
    -0.5791322f, -0.4022501f, -0.2372021f, -0.0784124f,
     0.0784124f,  0.2372021f,  0.4022501f,  0.5791322f,
     0.7764218f,  1.0099893f,  1.3180109f,  1.8627319f };

// boundaries = midpoints of adjacent centroids
__device__ __constant__ float kBound[15] = {
    -1.5903714f, -1.1640001f, -0.8932056f, -0.6777770f,
    -0.4906912f, -0.3197261f, -0.1578073f,  0.0000000f,
     0.1578073f,  0.3197261f,  0.4906912f,  0.6777770f,
     0.8932056f,  1.1640001f,  1.5903714f };

__device__ __forceinline__ unsigned short f2bf(float f) {
    unsigned int u = __builtin_bit_cast(unsigned int, f);
    unsigned int r = u + 0x7FFFu + ((u >> 16) & 1u);   // round-to-nearest-even
    return (unsigned short)(r >> 16);
}

// ---------------------------------------------------------------------------
// Kernel 1: per-128-block norm + FWHT + quantize, emit bf16 (centroid * scale)
// One wave32 per block: 4 elems/lane. Memory-bound streaming pass.
// ---------------------------------------------------------------------------
__global__ void __launch_bounds__(256)
had_quant_kernel(const float* __restrict__ x, unsigned short* __restrict__ xq,
                 int nblocks) {
    int wid  = (int)((blockIdx.x * blockDim.x + threadIdx.x) >> 5);
    int lane = threadIdx.x & 31;
    if (wid >= nblocks) return;

    const size_t base = (size_t)wid * BS + (size_t)lane * 4;
    float4 d = *(const float4*)(x + base);
    float v0 = d.x, v1 = d.y, v2 = d.z, v3 = d.w;

    // block L2 norm of the ORIGINAL data
    float ss = v0 * v0 + v1 * v1 + v2 * v2 + v3 * v3;
    #pragma unroll
    for (int m = 1; m < 32; m <<= 1) ss += __shfl_xor(ss, m, 32);
    float scale = sqrtf(ss);

    // FWHT, in-lane stages h=1,2 (elements e = 4*lane + i)
    float t0 = v0 + v1, t1 = v0 - v1, t2 = v2 + v3, t3 = v2 - v3;
    v0 = t0 + t2; v2 = t0 - t2; v1 = t1 + t3; v3 = t1 - t3;

    // cross-lane stages h=4..64  ->  lane-xor masks 1..16 (wave32 native)
    #pragma unroll
    for (int m = 1; m < 32; m <<= 1) {
        float p0 = __shfl_xor(v0, m, 32);
        float p1 = __shfl_xor(v1, m, 32);
        float p2 = __shfl_xor(v2, m, 32);
        float p3 = __shfl_xor(v3, m, 32);
        bool up = (lane & m) != 0;
        v0 = up ? (p0 - v0) : (v0 + p0);
        v1 = up ? (p1 - v1) : (v1 + p1);
        v2 = up ? (p2 - v2) : (v2 + p2);
        v3 = up ? (p3 - v3) : (v3 + p3);
    }

    float inv = (scale > 0.f) ? (1.0f / scale) : 0.f;
    float vv[4] = { v0, v1, v2, v3 };
    unsigned short o[4];
    #pragma unroll
    for (int i = 0; i < 4; ++i) {
        float u = vv[i] * inv;
        int idx = 0;
        #pragma unroll
        for (int j = 0; j < 15; ++j) idx += (u > kBound[j]) ? 1 : 0;
        o[i] = f2bf(kCentroids[idx] * scale);   // dequantized, pre-1/128 scaling
    }
    uint2 pk;
    pk.x = (unsigned int)o[0] | ((unsigned int)o[1] << 16);
    pk.y = (unsigned int)o[2] | ((unsigned int)o[3] << 16);
    *(uint2*)(xq + base) = pk;
}

// ---------------------------------------------------------------------------
// Kernel 2: weight f32 -> bf16 (RNE), streaming
// ---------------------------------------------------------------------------
__global__ void __launch_bounds__(256)
w_convert_kernel(const float* __restrict__ w, unsigned short* __restrict__ wb,
                 size_t n) {
    size_t i = ((size_t)blockIdx.x * blockDim.x + threadIdx.x) * 4;
    if (i >= n) return;
    float4 d = *(const float4*)(w + i);
    uint2 pk;
    pk.x = (unsigned int)f2bf(d.x) | ((unsigned int)f2bf(d.y) << 16);
    pk.y = (unsigned int)f2bf(d.z) | ((unsigned int)f2bf(d.w) << 16);
    *(uint2*)(wb + i) = pk;
}

// ---------------------------------------------------------------------------
// GEMM helpers
// ---------------------------------------------------------------------------
#if HAS_ASYNC
// Issue one tile's worth of ASYNCcnt-tracked copies straight into LDS:
// per thread 4x b128 for A (row tid>>1, 32-elem half) and 2x4 b128 for B
// (rows tid>>1 and tid>>1 + 128). 12 issues/thread -> ASYNCcnt += 12.
__device__ __forceinline__ void
fill_async(const unsigned short* A, const unsigned short* B,
           unsigned short* sA, unsigned short* sB,
           int bm, int bn, int k0, int tid) {
    const int row = tid >> 1;
    const int col = (tid & 1) * 32;

    {
        const unsigned short* g = A + (size_t)(bm + row) * K_DIM + k0 + col;
        unsigned short*       l = sA + row * APITCH + col;
        AS1 vint4* gp = (AS1 vint4*)(uintptr_t)g;
        AS3 vint4* lp = (AS3 vint4*)(uintptr_t)l;   // low 32 bits = LDS offset
        __builtin_amdgcn_global_load_async_to_lds_b128(gp, lp, 0, 0);
        __builtin_amdgcn_global_load_async_to_lds_b128(gp, lp, 16, 0);
        __builtin_amdgcn_global_load_async_to_lds_b128(gp, lp, 32, 0);
        __builtin_amdgcn_global_load_async_to_lds_b128(gp, lp, 48, 0);
    }
    #pragma unroll
    for (int grp = 0; grp < 2; ++grp) {
        const int r = row + grp * 128;
        const unsigned short* g = B + (size_t)(bn + r) * K_DIM + k0 + col;
        unsigned short*       l = sB + r * APITCH + col;
        AS1 vint4* gp = (AS1 vint4*)(uintptr_t)g;
        AS3 vint4* lp = (AS3 vint4*)(uintptr_t)l;
        __builtin_amdgcn_global_load_async_to_lds_b128(gp, lp, 0, 0);
        __builtin_amdgcn_global_load_async_to_lds_b128(gp, lp, 16, 0);
        __builtin_amdgcn_global_load_async_to_lds_b128(gp, lp, 32, 0);
        __builtin_amdgcn_global_load_async_to_lds_b128(gp, lp, 48, 0);
    }
}
#else
// Fallback: global -> regs -> ds_store (single-buffered, caller barriers)
__device__ __forceinline__ void
fill_sync(const unsigned short* A, const unsigned short* B,
          unsigned short* sA, unsigned short* sB,
          int bm, int bn, int k0, int tid) {
    const int row = tid >> 1;
    const int col = (tid & 1) * 32;
    const uint4* ga = (const uint4*)(A + (size_t)(bm + row) * K_DIM + k0 + col);
    #pragma unroll
    for (int i = 0; i < 4; ++i)
        *(uint4*)&sA[row * APITCH + col + i * 8] = ga[i];
    #pragma unroll
    for (int grp = 0; grp < 2; ++grp) {
        const int r = row + grp * 128;
        const uint4* gb = (const uint4*)(B + (size_t)(bn + r) * K_DIM + k0 + col);
        #pragma unroll
        for (int i = 0; i < 4; ++i)
            *(uint4*)&sB[r * APITCH + col + i * 8] = gb[i];
    }
}
#endif

// One K-tile of WMMAs: 2 k-steps x (4 A-frags, 4 B-frags, 16 WMMA)
__device__ __forceinline__ void
compute_tile(const unsigned short* sA, const unsigned short* sB,
             v8f acc[4][4], int wm, int wn, int lane) {
    const int hi  = lane >> 4;    // K-half select per ISA 16-bit fragment map
    const int l15 = lane & 15;
    #pragma unroll
    for (int ks = 0; ks < 2; ++ks) {
        v16bf af[4], bfr[4];
        #pragma unroll
        for (int mt = 0; mt < 4; ++mt) {
            // A 16x32 bf16: lanes0-15 K0..7|K16..23 ; lanes16-31 K8..15|K24..31
            const unsigned short* ap =
                &sA[(wm * 64 + mt * 16 + l15) * APITCH + ks * 32 + hi * 8];
            union { uint4 u[2]; v16bf v; } t;
            t.u[0] = *(const uint4*)(ap);
            t.u[1] = *(const uint4*)(ap + 16);
            af[mt] = t.v;
        }
        #pragma unroll
        for (int nt = 0; nt < 4; ++nt) {
            // B 32x16 bf16: lanes0-15 col N, K0..7|K16..23 ; lanes16-31 K8..15|K24..31
            const unsigned short* bp =
                &sB[(wn * 64 + nt * 16 + l15) * APITCH + ks * 32 + hi * 8];
            union { uint4 u[2]; v16bf v; } t;
            t.u[0] = *(const uint4*)(bp);
            t.u[1] = *(const uint4*)(bp + 16);
            bfr[nt] = t.v;
        }
        #pragma unroll
        for (int mt = 0; mt < 4; ++mt)
            #pragma unroll
            for (int nt = 0; nt < 4; ++nt)
                acc[mt][nt] = __builtin_amdgcn_wmma_f32_16x16x32_bf16(
                    false, af[mt], false, bfr[nt],
                    (short)0, acc[mt][nt], false, false);
    }
}

// ---------------------------------------------------------------------------
// Kernel 3: C[M,N] = (A[M,K] @ B[N,K]^T) / 128   (A,B bf16, C f32)
// Block tile 128x256, 8 wave32s each 64x64 (4x4 WMMA 16x16x32),
// double-buffered LDS fed by ASYNCcnt-tracked global_load_async_to_lds_b128.
// ---------------------------------------------------------------------------
__global__ void __launch_bounds__(256)
gemm_kernel(const unsigned short* __restrict__ A,   // M x K bf16 row-major
            const unsigned short* __restrict__ B,   // N x K bf16 row-major
            float* __restrict__ C, int M) {
    __shared__ unsigned short sA[NBUF][BM * APITCH];
    __shared__ unsigned short sB[NBUF][BN * APITCH];

    const int tid  = threadIdx.x;
    const int lane = tid & 31;
    const int wave = tid >> 5;
    const int wm   = wave & 1;   // 0..1 : 64-row half
    const int wn   = wave >> 1;  // 0..3 : 64-col quarter
    const int bm   = blockIdx.y * BM;
    const int bn   = blockIdx.x * BN;

    v8f acc[4][4];
    #pragma unroll
    for (int i = 0; i < 4; ++i)
        #pragma unroll
        for (int j = 0; j < 4; ++j)
            acc[i][j] = (v8f)(0.0f);

    const int NK = K_DIM / BK;

#if HAS_ASYNC
    fill_async(A, B, sA[0], sB[0], bm, bn, 0, tid);
    for (int kt = 0; kt < NK; ++kt) {
        const int cur = kt & 1;
        if (kt + 1 < NK) {
            fill_async(A, B, sA[cur ^ 1], sB[cur ^ 1], bm, bn, (kt + 1) * BK, tid);
            __builtin_amdgcn_s_wait_asynccnt(12);  // current tile's 12 done (FIFO)
        } else {
            __builtin_amdgcn_s_wait_asynccnt(0);
        }
        __syncthreads();                           // tile visible to all waves
        compute_tile(sA[cur], sB[cur], acc, wm, wn, lane);
        __syncthreads();                           // done reading before overwrite
    }
#else
    for (int kt = 0; kt < NK; ++kt) {
        __syncthreads();
        fill_sync(A, B, sA[0], sB[0], bm, bn, kt * BK, tid);
        __syncthreads();
        compute_tile(sA[0], sB[0], acc, wm, wn, lane);
    }
#endif

    // epilogue: C/D 16x16 f32 layout — lane 0-15: N=lane, VGPR r: M=r;
    //           lane 16-31: N=lane-16, VGPR r: M=8+r
    const int hi  = lane >> 4;
    const int l15 = lane & 15;
    const float s = 1.0f / 128.0f;   // the two 1/sqrt(128) factors
    #pragma unroll
    for (int mt = 0; mt < 4; ++mt) {
        #pragma unroll
        for (int nt = 0; nt < 4; ++nt) {
            int col     = bn + wn * 64 + nt * 16 + l15;
            int rowbase = bm + wm * 64 + mt * 16 + hi * 8;
            #pragma unroll
            for (int r = 0; r < 8; ++r) {
                C[(size_t)(rowbase + r) * N_DIM + col] = acc[mt][nt][r] * s;
            }
        }
    }
}

// ---------------------------------------------------------------------------
extern "C" void kernel_launch(void* const* d_in, const int* in_sizes, int n_in,
                              void* d_out, int out_size, void* d_ws, size_t ws_size,
                              hipStream_t stream) {
    const float* x = (const float*)d_in[0];   // (4,4096,4096) f32
    const float* w = (const float*)d_in[1];   // (4096,4096) f32
    float* out = (float*)d_out;               // (4,4096,4096) f32

    const size_t xelems = (size_t)in_sizes[0];
    const size_t welems = (size_t)in_sizes[1];
    const int M = (int)(xelems / K_DIM);      // 16384

    unsigned short* xq = (unsigned short*)d_ws;        // M*K bf16 (128 MB)
    unsigned short* wb = xq + xelems;                  // N*K bf16 (32 MB)

    // 1) FWHT + quantize
    const int nblocks = (int)(xelems / BS);            // one wave32 per block
    const int qgrid   = (nblocks + 7) / 8;             // 8 waves / 256-thread WG
    hipLaunchKernelGGL(had_quant_kernel, dim3(qgrid), dim3(256), 0, stream,
                       x, xq, nblocks);

    // 2) weight -> bf16
    const int wthreads = (int)(welems / 4);
    hipLaunchKernelGGL(w_convert_kernel, dim3((wthreads + 255) / 256), dim3(256),
                       0, stream, w, wb, welems);

    // 3) WMMA GEMM with 1/128 epilogue scale
    dim3 grid(N_DIM / BN, M / BM);
    hipLaunchKernelGGL(gemm_kernel, grid, dim3(256), 0, stream, xq, wb, out, M);
}
